// SimpleNet_83837761618434
// MI455X (gfx1250) — compile-verified, matrix operand order checked by reading, not run on hardware
//
#include <hip/hip_runtime.h>
#include <hip/hip_bf16.h>
#include <math.h>

#define N_NODES 100000
#define N_EDGES 1600000
#define DIM_IN  165
#define DIM_HID 128
#define KPAD    192   // DIM_IN padded up to a multiple of 32 (WMMA K)
#define NCAT    256   // [W_rel1 | W_root1] stacked: 2*DIM_HID output channels

typedef __attribute__((ext_vector_type(16))) __bf16 v16bf;
typedef __attribute__((ext_vector_type(8)))  __bf16 v8bf;
typedef __attribute__((ext_vector_type(8)))  float  v8f;

// ---------------------------------------------------------------- zero fill
__global__ void k_zero(float* __restrict__ p, int n) {
    int i = blockIdx.x * blockDim.x + threadIdx.x;
    if (i < n) p[i] = 0.0f;
}

// ------------------------------------------- pack [W_rel1;W_root1] -> bf16
// Wb layout: [NCAT][KPAD] row-major, K padded with zeros (165..191).
__global__ void k_pack_w(const float* __restrict__ Wrel,
                         const float* __restrict__ Wroot,
                         __bf16* __restrict__ Wb) {
    int i = blockIdx.x * blockDim.x + threadIdx.x;
    if (i >= NCAT * KPAD) return;
    int n = i / KPAD, k = i % KPAD;
    float v = 0.0f;
    if (k < DIM_IN)
        v = (n < DIM_HID) ? Wrel[n * DIM_IN + k] : Wroot[(n - DIM_HID) * DIM_IN + k];
    Wb[i] = (__bf16)v;
}

// ------------------------------------------------------- WMMA GEMM kernel
// ycat[node][0:128]   = x @ W_rel1^T   (to be edge-aggregated)
// ycat[node][128:256] = x @ W_root1^T  (self term)
// Block: 256 threads = 8 waves, tiled 4(M) x 2(N): 64 rows x 256 cols/block.
// Each wave: 16 rows x 128 cols = 8 C-tiles, K-loop of 6 -> 48 v_wmma.
__global__ void __launch_bounds__(256)
k_gemm(const float* __restrict__ x,
       const __bf16* __restrict__ Wb,
       float* __restrict__ ycat) {
    const int lane    = threadIdx.x & 31;
    const int wave    = threadIdx.x >> 5;
    const int mw      = wave & 3;       // M sub-tile 0..3
    const int nw      = wave >> 2;      // N half 0..1
    const int m0      = blockIdx.x * 64 + mw * 16;
    const int n0      = nw * 128;
    const int halfSel = lane >> 4;      // 0: lanes 0-15, 1: lanes 16-31
    const int nlane   = lane & 15;      // M index for A, N index for B/C

    // A-fragment source row (clamped; stores are guarded instead)
    const int m  = m0 + nlane;
    const int mc = (m < N_NODES) ? m : (N_NODES - 1);
    const float* xr = x + (size_t)mc * DIM_IN;

    v8f acc[8] = {};

    #pragma unroll
    for (int ks = 0; ks < 6; ++ks) {
        const int kk    = ks * 32;
        const int kbase = kk + halfSel * 8;

        // Build bf16 A fragment per ISA 16-bit 16x32 layout:
        // lane L: M = L%16; halves 0..7 -> K=kbase..kbase+7, 8..15 -> K=kbase+16..+23
        v16bf a;
        #pragma unroll
        for (int h = 0; h < 8; ++h) {
            int k0 = kbase + h;
            int k1 = kbase + 16 + h;
            a[h]     = (__bf16)((k0 < DIM_IN) ? xr[k0] : 0.0f);
            a[h + 8] = (__bf16)((k1 < DIM_IN) ? xr[k1] : 0.0f);
        }

        #pragma unroll
        for (int t = 0; t < 8; ++t) {
            // B fragment: lane L holds column n = n0+16t+(L%16), same K split.
            const __bf16* wrow = Wb + (size_t)(n0 + t * 16 + nlane) * KPAD + kbase;
            v8bf blo = *(const v8bf*)(wrow);        // K = kbase..kbase+7   (16B aligned)
            v8bf bhi = *(const v8bf*)(wrow + 16);   // K = kbase+16..+23
            v16bf b = __builtin_shufflevector(blo, bhi,
                        0,1,2,3,4,5,6,7,8,9,10,11,12,13,14,15);
            acc[t] = __builtin_amdgcn_wmma_f32_16x16x32_bf16(
                        false, a, false, b, (short)0, acc[t], false, false);
        }
    }

    // C layout: lane L -> N = L%16; VGPR r -> M = r + 8*(L>=16)
    #pragma unroll
    for (int t = 0; t < 8; ++t) {
        const int col = n0 + t * 16 + nlane;
        #pragma unroll
        for (int r = 0; r < 8; ++r) {
            const int row = m0 + r + 8 * halfSel;
            if (row < N_NODES)
                ycat[(size_t)row * NCAT + col] = acc[t][r];
        }
    }
}

// -------------------------------------- layer-1 edge aggregation (128-dim)
// wave per edge; lane handles 4 channels: agg[dst] += y_rel[src]
__global__ void __launch_bounds__(256)
k_agg1(const int* __restrict__ src, const int* __restrict__ dst,
       const float* __restrict__ ycat, float* __restrict__ agg) {
    const int lane = threadIdx.x & 31;
    const int wave = threadIdx.x >> 5;
    const int e    = blockIdx.x * 8 + wave;
    if (e >= N_EDGES) return;
    const int s = src[e];
    const int d = dst[e];
    const float4 v = *(const float4*)(ycat + (size_t)s * NCAT + lane * 4);
    float* ar = agg + (size_t)d * DIM_HID + lane * 4;
    atomicAdd(ar + 0, v.x);
    atomicAdd(ar + 1, v.y);
    atomicAdd(ar + 2, v.z);
    atomicAdd(ar + 3, v.w);
}

// ----------------------- layer-1 epilogue + layer-2 node-side dot products
// wave per node: h = relu(agg + y_root + b1); s_rel = h.W_rel2; s_root = h.W_root2
__global__ void __launch_bounds__(256)
k_node2(const float* __restrict__ agg, const float* __restrict__ ycat,
        const float* __restrict__ b_rel1,
        const float* __restrict__ W_rel2, const float* __restrict__ W_root2,
        float* __restrict__ s_rel, float* __restrict__ s_root) {
    const int lane = threadIdx.x & 31;
    const int wave = threadIdx.x >> 5;
    const int i    = blockIdx.x * 8 + wave;
    if (i >= N_NODES) return;
    const int c = lane * 4;

    float4 a  = *(const float4*)(agg   + (size_t)i * DIM_HID + c);
    float4 yr = *(const float4*)(ycat  + (size_t)i * NCAT + DIM_HID + c);
    float4 bb = *(const float4*)(b_rel1 + c);
    float4 w1 = *(const float4*)(W_rel2 + c);
    float4 w2 = *(const float4*)(W_root2 + c);

    float h0 = fmaxf(a.x + yr.x + bb.x, 0.0f);
    float h1 = fmaxf(a.y + yr.y + bb.y, 0.0f);
    float h2 = fmaxf(a.z + yr.z + bb.z, 0.0f);
    float h3 = fmaxf(a.w + yr.w + bb.w, 0.0f);

    float sr = h0 * w1.x + h1 * w1.y + h2 * w1.z + h3 * w1.w;
    float so = h0 * w2.x + h1 * w2.y + h2 * w2.z + h3 * w2.w;

    #pragma unroll
    for (int off = 16; off > 0; off >>= 1) {
        sr += __shfl_xor(sr, off, 32);
        so += __shfl_xor(so, off, 32);
    }
    if (lane == 0) { s_rel[i] = sr; s_root[i] = so; }
}

// ------------------------------------- layer-2 edge aggregation (scalars)
__global__ void k_agg2(const int* __restrict__ src, const int* __restrict__ dst,
                       const float* __restrict__ s_rel, float* __restrict__ acc2) {
    int e = blockIdx.x * blockDim.x + threadIdx.x;
    if (e < N_EDGES) atomicAdd(&acc2[dst[e]], s_rel[src[e]]);
}

// ------------------------------------------------------------ final sigmoid
__global__ void k_final(const float* __restrict__ acc2, const float* __restrict__ s_root,
                        const float* __restrict__ b_rel2, float* __restrict__ out) {
    int i = blockIdx.x * blockDim.x + threadIdx.x;
    if (i < N_NODES) {
        float z = acc2[i] + s_root[i] + b_rel2[0];
        out[i] = 1.0f / (1.0f + expf(-z));
    }
}

extern "C" void kernel_launch(void* const* d_in, const int* in_sizes, int n_in,
                              void* d_out, int out_size, void* d_ws, size_t ws_size,
                              hipStream_t stream) {
    const float* x       = (const float*)d_in[0];
    const int*   edges   = (const int*)d_in[1];   // [2][N_EDGES], int32
    const float* W_rel1  = (const float*)d_in[2];
    const float* b_rel1  = (const float*)d_in[3];
    const float* W_root1 = (const float*)d_in[4];
    const float* W_rel2  = (const float*)d_in[5];
    const float* b_rel2  = (const float*)d_in[6];
    const float* W_root2 = (const float*)d_in[7];
    const int* src = edges;
    const int* dst = edges + N_EDGES;
    float* out = (float*)d_out;

    // Workspace layout (~155 MB total)
    char* ws = (char*)d_ws;
    float*  ycat   = (float*)(ws);                 // 100000*256 f32 = 102.4 MB
    float*  agg    = (float*)(ws + 102400000);     // 100000*128 f32 =  51.2 MB
    float*  s_rel  = (float*)(ws + 153600000);     // 100000 f32
    float*  s_root = (float*)(ws + 154000000);     // 100000 f32
    float*  acc2   = (float*)(ws + 154400000);     // 100000 f32
    __bf16* Wb     = (__bf16*)(ws + 154800000);    // 256*192 bf16 = 96 KB

    k_zero  <<<(N_NODES * DIM_HID + 255) / 256, 256, 0, stream>>>(agg, N_NODES * DIM_HID);
    k_zero  <<<(N_NODES + 255) / 256,           256, 0, stream>>>(acc2, N_NODES);
    k_pack_w<<<(NCAT * KPAD + 255) / 256,       256, 0, stream>>>(W_rel1, W_root1, Wb);
    k_gemm  <<<(N_NODES + 63) / 64,             256, 0, stream>>>(x, Wb, ycat);
    k_agg1  <<<N_EDGES / 8,                     256, 0, stream>>>(src, dst, ycat, agg);
    k_node2 <<<(N_NODES + 7) / 8,               256, 0, stream>>>(agg, ycat, b_rel1, W_rel2, W_root2, s_rel, s_root);
    k_agg2  <<<(N_EDGES + 255) / 256,           256, 0, stream>>>(src, dst, s_rel, acc2);
    k_final <<<(N_NODES + 255) / 256,           256, 0, stream>>>(acc2, s_root, b_rel2, out);
}
